// TernaryLinear_36369783063105
// MI455X (gfx1250) — matrix operand order repaired and checked by reading, not compiled, not run on hardware
//
#include <hip/hip_runtime.h>
#include <hip/hip_bf16.h>
#include <stdint.h>

// ---------------------------------------------------------------------------
// TernaryLinear train-forward for MI455X (gfx1250, wave32, WMMA).
//   out = (x @ Wm^T + sqrt((x*x) @ Wv^T) * noise) * scale + shift
// Both GEMMs run fused in one kernel, sharing the A operand (x / x^2).
// B operands are staged through LDS with gfx1250 async global->LDS copies.
// ---------------------------------------------------------------------------

#define IN_DIMS   1024
#define OUT_DIMS  1024
#define N_TOKENS  32768

typedef __attribute__((ext_vector_type(16))) _Float16 v16h;
typedef __attribute__((ext_vector_type(8)))  float    v8f;
typedef __attribute__((ext_vector_type(4)))  float    v4f;

constexpr int K_TILES = IN_DIMS / 32;   // 32 K-tiles of 32
constexpr int O_TILES = OUT_DIMS / 16;  // 64 O-tiles of 16

// ---------------- gfx1250 async copy helpers --------------------------------
// global_load_async_to_lds_b128: per-lane copy of 16B global -> 16B LDS.
// VDST = LDS byte address (low 32 bits of generic LDS pointer), VADDR = VA.
__device__ inline void async_copy_b128(uint32_t lds_addr, const void* gaddr) {
    asm volatile("global_load_async_to_lds_b128 %0, %1, off"
                 :: "v"(lds_addr), "v"(gaddr)
                 : "memory");
}

__device__ inline void wait_async_le4() {
#if __has_builtin(__builtin_amdgcn_s_wait_asynccnt)
    __builtin_amdgcn_s_wait_asynccnt(4);
#else
    asm volatile("s_wait_asynccnt 0x4" ::: "memory");
#endif
}
__device__ inline void wait_async_le0() {
#if __has_builtin(__builtin_amdgcn_s_wait_asynccnt)
    __builtin_amdgcn_s_wait_asynccnt(0);
#else
    asm volatile("s_wait_asynccnt 0x0" ::: "memory");
#endif
}

// ---------------------------------------------------------------------------
// Kernel 1: weight transform. sigmoid(logits) -> ternary mean/var, stored in
// f16 directly in WMMA B-fragment blocked layout:
//   fragment block b = o_tile * K_TILES + k_tile  (16 o-cols x 32 k-rows)
//   lane l owns column o = o_tile*16 + (l & 15),
//             K-run   k = k_tile*32 + (l >> 4)*16 + e,  e = 0..15
//   storage: contiguous 16 halves per lane at (b*32 + l)*16.
// One wave handles one fragment block (512 elements). 4 MB total -> L2.
// ---------------------------------------------------------------------------
__global__ __launch_bounds__(256) void ternary_prep_weights(
    const float* __restrict__ wl,   // [O, I, 2] (mag_logit, sgn_logit)
    _Float16* __restrict__ wm,      // [O_TILES*K_TILES*512] blocked
    _Float16* __restrict__ wv)      // same layout
{
    const int lane = threadIdx.x & 31;
    const int wave = threadIdx.x >> 5;
    const int blk  = blockIdx.x * 8 + wave;      // fragment-block index
    const int ot   = blk / K_TILES;
    const int kt   = blk - ot * K_TILES;
    const int o     = ot * 16 + (lane & 15);
    const int kbase = kt * 32 + (lane >> 4) * 16;

    v16h m16, v16;
#pragma unroll
    for (int e = 0; e < 16; ++e) {
        const int k = kbase + e;
        const float2 z = *(const float2*)(wl + ((size_t)o * IN_DIMS + k) * 2);
        const float pm = 1.0f / (1.0f + __expf(-z.x));   // P(|w| = 1)
        const float ps = 1.0f / (1.0f + __expf(-z.y));   // P(w > 0)
        const float s  = 2.0f * ps - 1.0f;
        const float mean = pm * s;
        const float var  = pm - (pm * pm) * (s * s);
        m16[e] = (_Float16)mean;
        v16[e] = (_Float16)var;
    }
    const size_t base = ((size_t)blk * 32 + lane) * 16;
    *(v16h*)(wm + base) = m16;
    *(v16h*)(wv + base) = v16;
}

// ---------------------------------------------------------------------------
// Kernel 2: fused dual GEMM + epilogue.
// Block: 256 threads = 8 waves, tile 128(N) x 128(O).
// Wave grid 4(M) x 2(O): each wave owns a 32x64 tile = 2x4 WMMA tiles,
// each with two f32 accumulators (mean, var) -> 16 WMMAs per K-step.
//
// B staging: per K-step the block needs 16 chunks x 1KB of B fragments
// (8 o-blocks x {Wm,Wv}), shared by all waves. These are copied with
// async global->LDS b128 ops into a double buffer (32 KB LDS) and read
// back as 32B/lane ds_load fragments. ASYNCcnt pipelining: stage K+1,
// s_wait_asynccnt 4 retires exactly the current buffer's 4 per-thread
// copies (async loads complete in order).
// ---------------------------------------------------------------------------
constexpr int STAGE_HALVES = 16 * 512;          // 16 KB per buffer

__device__ inline void stage_B(const _Float16* __restrict__ wm,
                               const _Float16* __restrict__ wv,
                               _Float16* sbuf, int oblBase, int ktile, int tid) {
    const int c   = tid >> 4;                   // chunk 0..15 (1 KB each)
    const int t   = tid & 15;                   // 16 threads per chunk
    const int obl = c & 7;                      // o-fragment block in tile
    const _Float16* src = ((c < 8) ? wm : wv)
        + (((size_t)(oblBase + obl) * K_TILES + ktile) * 512);
    _Float16* dst = sbuf + (size_t)c * 512;
    const uint32_t dbase = (uint32_t)(uintptr_t)(dst + t * 32);  // 64B/thread
    const char* s = (const char*)(src + t * 32);
#pragma unroll
    for (int i = 0; i < 4; ++i)
        async_copy_b128(dbase + i * 16, s + i * 16);
}

__global__ __launch_bounds__(256) void ternary_gemm(
    const float* __restrict__ x,        // [N, I] f32
    const _Float16* __restrict__ wm,    // blocked B fragments
    const _Float16* __restrict__ wv,    // blocked B fragments
    const float* __restrict__ scale,    // [O]
    const float* __restrict__ shift,    // [O]
    const float* __restrict__ noise,    // [N, O]
    float* __restrict__ out)            // [N, O]
{
    __shared__ _Float16 sB[2 * STAGE_HALVES];   // 32 KB double buffer

    const int tid  = threadIdx.x;
    const int lane = tid & 31;
    const int wave = tid >> 5;
    const int wrow = wave & 3;          // 4 waves along N
    const int wcol = wave >> 2;         // 2 waves along O
    const int waveM = blockIdx.x * 128 + wrow * 32;
    const int waveO = blockIdx.y * 128 + wcol * 64;
    const int lrow = lane & 15;
    const int lhi  = lane >> 4;
    const int oblBase = blockIdx.y * 8;

    const v8f zero8 = {0.f, 0.f, 0.f, 0.f, 0.f, 0.f, 0.f, 0.f};
    v8f accM[2][4], accV[2][4];
#pragma unroll
    for (int mi = 0; mi < 2; ++mi)
#pragma unroll
        for (int oi = 0; oi < 4; ++oi) { accM[mi][oi] = zero8; accV[mi][oi] = zero8; }

    // prologue: stage K-tile 0
    stage_B(wm, wv, sB, oblBase, 0, tid);

    for (int ktile = 0; ktile < K_TILES; ++ktile) {
        const int kk = ktile * 32;

        if (ktile + 1 < K_TILES) {
            stage_B(wm, wv, sB + ((ktile + 1) & 1) * STAGE_HALVES,
                    oblBase, ktile + 1, tid);
            wait_async_le4();           // current buffer's copies retired
        } else {
            wait_async_le0();
        }
        __syncthreads();                // staged B visible to all waves

        // ---- A fragments: x (f32 -> f16), then x^2 in packed f16 ----
        v16h ax[2], ax2[2];
#pragma unroll
        for (int mi = 0; mi < 2; ++mi) {
            const float* px = x + (size_t)(waveM + mi * 16 + lrow) * IN_DIMS
                                + kk + lhi * 16;
            const v4f p0 = *(const v4f*)(px);
            const v4f p1 = *(const v4f*)(px + 4);
            const v4f p2 = *(const v4f*)(px + 8);
            const v4f p3 = *(const v4f*)(px + 12);
            if (kk + 32 < IN_DIMS)
                __builtin_prefetch(px + 32, 0, 0);   // next K-step of this row
#pragma unroll
            for (int j = 0; j < 4; ++j) {
                ax[mi][j]      = (_Float16)p0[j];
                ax[mi][4 + j]  = (_Float16)p1[j];
                ax[mi][8 + j]  = (_Float16)p2[j];
                ax[mi][12 + j] = (_Float16)p3[j];
            }
            ax2[mi] = ax[mi] * ax[mi];          // v_pk_mul_f16 x8
        }

        // ---- B fragments from LDS: 32B/lane ds loads ----
        const _Float16* buf = sB + (ktile & 1) * STAGE_HALVES;
        v16h bm[4], bv[4];
#pragma unroll
        for (int oi = 0; oi < 4; ++oi) {
            const int obl = wcol * 4 + oi;
            bm[oi] = *(const v16h*)(buf + (size_t)obl * 512 + lane * 16);
            bv[oi] = *(const v16h*)(buf + (size_t)(8 + obl) * 512 + lane * 16);
        }

        // ---- 16 WMMAs: dual GEMM on the matrix pipes ----
#pragma unroll
        for (int mi = 0; mi < 2; ++mi)
#pragma unroll
            for (int oi = 0; oi < 4; ++oi) {
                accM[mi][oi] = __builtin_amdgcn_wmma_f32_16x16x32_f16(
                    false, ax[mi],  false, bm[oi], (short)0, accM[mi][oi],
                    false, false);
                accV[mi][oi] = __builtin_amdgcn_wmma_f32_16x16x32_f16(
                    false, ax2[mi], false, bv[oi], (short)0, accV[mi][oi],
                    false, false);
            }

        __syncthreads();                // buffer may be overwritten next iter
    }

    // ---- Epilogue: out = (mean + sqrt(max(var,0)) * noise) * scale + shift ----
    // C/D layout: VGPR r, lane l -> n_local = (l>>4)*8 + r, o_local = l&15.
#pragma unroll
    for (int oi = 0; oi < 4; ++oi) {
        const int o  = waveO + oi * 16 + lrow;
        const float sc = scale[o];
        const float sh = shift[o];
#pragma unroll
        for (int mi = 0; mi < 2; ++mi) {
#pragma unroll
            for (int r = 0; r < 8; ++r) {
                const int n = waveM + mi * 16 + lhi * 8 + r;
                const float mean = accM[mi][oi][r];
                float var = accV[mi][oi][r];
                var = var > 0.0f ? var : 0.0f;
                const float nz = noise[(size_t)n * OUT_DIMS + o];
                out[(size_t)n * OUT_DIMS + o] =
                    (mean + __builtin_sqrtf(var) * nz) * sc + sh;
            }
        }
    }
}

// ---------------------------------------------------------------------------
// Host launch. Inputs (setup_inputs order):
//   d_in[0] = x [N, I] f32
//   d_in[1] = weight_logits [O, I, 2] f32
//   d_in[2] = scale [O] f32
//   d_in[3] = shift [O] f32
//   d_in[4] = noise [N, O] f32
// d_ws: 4 MB used (Wm, Wv f16 blocked fragments).
// ---------------------------------------------------------------------------
extern "C" void kernel_launch(void* const* d_in, const int* in_sizes, int n_in,
                              void* d_out, int out_size, void* d_ws, size_t ws_size,
                              hipStream_t stream) {
    const float* x     = (const float*)d_in[0];
    const float* wl    = (const float*)d_in[1];
    const float* scale = (const float*)d_in[2];
    const float* shift = (const float*)d_in[3];
    const float* noise = (const float*)d_in[4];
    float* out = (float*)d_out;

    _Float16* wm = (_Float16*)d_ws;
    _Float16* wv = wm + (size_t)OUT_DIMS * IN_DIMS;

    // 2048 fragment blocks, 8 per 256-thread block
    ternary_prep_weights<<<(O_TILES * K_TILES) / 8, 256, 0, stream>>>(wl, wm, wv);

    dim3 grid(N_TOKENS / 128, OUT_DIMS / 128);
    ternary_gemm<<<grid, 256, 0, stream>>>(x, wm, wv, scale, shift, noise, out);
}
